// PointnetSAModuleVotes_16260746183080
// MI455X (gfx1250) — matrix operand order repaired and checked by reading, not compiled, not run on hardware
//
#include <hip/hip_runtime.h>

// ---------------------------------------------------------------------------
// Problem constants (match the JAX reference)
// ---------------------------------------------------------------------------
#define BB      8
#define NN      8192
#define SS      2048          // NPOINT
#define NS      64            // NSAMPLE
#define CIN     64
#define RADIUS  0.2f
#define R2      (RADIUS * RADIUS)

// WMMA fragment types (CDNA5 wave32, 16x16x32 f16 -> f32)
typedef __attribute__((ext_vector_type(16))) _Float16 hvec16;
typedef __attribute__((ext_vector_type(8)))  _Float16 hvec8;
typedef __attribute__((ext_vector_type(8)))  float    fvec8;

#define WMMA16(A, B, C) \
  __builtin_amdgcn_wmma_f32_16x16x32_f16(false, (A), false, (B), (short)0, (C), false, false)

// ---------------------------------------------------------------------------
// Fragment loaders (wave32; A and B share one contiguous K order)
//   A (16x32 f16): lane l, M = l&15 ; halves [0..7] = K kb..kb+7,
//                  halves [8..15] = K kb+8..kb+15, kb = 16*(l>>4)
//   B (32x16 f16): lane l, N = l&15 ; halves [0..15] = K contiguous,
//                  base K = 16*(l>>4)  (weights stored N-major so contiguous)
//   C/D (16x16 f32): VGPR r, lane l -> M = r + 8*(l>>4), N = l&15
// ---------------------------------------------------------------------------
__device__ __forceinline__ hvec16 load_a_frag(const _Float16* X, int ld,
                                              int mt, int kt, int lane) {
  const int m  = mt * 16 + (lane & 15);
  const int kb = kt * 32 + ((lane >> 4) << 4);
  const _Float16* p = X + m * ld + kb;
  hvec8 lo = *(const hvec8*)(p);        // K = kb .. kb+7
  hvec8 hi = *(const hvec8*)(p + 8);    // K = kb+8 .. kb+15
  hvec16 a;
#pragma unroll
  for (int e = 0; e < 8; ++e) { a[e] = lo[e]; a[e + 8] = hi[e]; }
  return a;
}

__device__ __forceinline__ hvec16 load_b_frag(const _Float16* WT, int ldk,
                                              int nt, int kt, int lane) {
  const int n  = nt * 16 + (lane & 15);
  const int kb = kt * 32 + ((lane >> 4) << 4);
  return *(const hvec16*)(WT + n * ldk + kb);
}

__device__ __forceinline__ void store_tile_relu(fvec8 acc, float sc, float sh,
                                                _Float16* Y, int ld,
                                                int mt, int nt, int lane) {
  const int col   = nt * 16 + (lane & 15);
  const int rbase = mt * 16 + ((lane >> 4) << 3);
#pragma unroll
  for (int r = 0; r < 8; ++r) {
    float v = fmaxf(fmaf(acc[r], sc, sh), 0.0f);
    Y[(rbase + r) * ld + col] = (_Float16)v;
  }
}

// ---------------------------------------------------------------------------
// Kernel 0: convert weights f32 -> f16, transposed to N-major, K padded.
//   w0 (67x64)  -> w0t[n*96 + k], n<64, k<96 (zero-pad k>=67)
//   w1 (64x64)  -> w1t[n*64 + k]
//   w2 (64x128) -> w2t[n*64 + k], n<128
// ---------------------------------------------------------------------------
__global__ void prep_weights_kernel(const float* __restrict__ w0,
                                    const float* __restrict__ w1,
                                    const float* __restrict__ w2,
                                    _Float16* __restrict__ w0t,
                                    _Float16* __restrict__ w1t,
                                    _Float16* __restrict__ w2t) {
  int t = blockIdx.x * blockDim.x + threadIdx.x;   // 18432 total
  if (t < 6144) {
    int n = t / 96, k = t % 96;
    w0t[t] = (k < 67) ? (_Float16)w0[k * 64 + n] : (_Float16)0.0f;
  } else if (t < 10240) {
    int i = t - 6144;
    int n = i / 64, k = i % 64;
    w1t[i] = (_Float16)w1[k * 64 + n];
  } else if (t < 18432) {
    int i = t - 10240;
    int n = i / 64, k = i % 64;
    w2t[i] = (_Float16)w2[k * 128 + n];
  }
}

// ---------------------------------------------------------------------------
// Kernel 1: gather new_xyz = xyz[inds]; emit new_xyz + inds outputs.
// ---------------------------------------------------------------------------
__global__ void gather_setup_kernel(const float* __restrict__ xyz,
                                    const int* __restrict__ inds,
                                    float* __restrict__ nxyz_ws,
                                    float* __restrict__ out_nxyz,
                                    float* __restrict__ out_inds) {
  int q = blockIdx.x * blockDim.x + threadIdx.x;
  if (q >= BB * SS) return;
  int b = q / SS;
  int j = inds[q];
  const float* p = xyz + ((size_t)b * NN + j) * 3;
#pragma unroll
  for (int c = 0; c < 3; ++c) {
    float v = p[c];
    nxyz_ws[q * 3 + c] = v;
    out_nxyz[q * 3 + c] = v;
  }
  out_inds[q] = (float)j;
}

// ---------------------------------------------------------------------------
// Kernel 2: ball query. One wave (32 lanes) per query, 8 queries per block.
// xyz staged through LDS in 1024-point (12 KB) chunks; workgroup-wide early
// exit once all 8 waves have 64 hits. Wave32 ballot + prefix-popc emits hit
// indices in ascending point order (CUDA ball_query semantics).
// ---------------------------------------------------------------------------
#define BQ_CHUNK 1024
__global__ void ball_query_kernel(const float* __restrict__ xyz,
                                  const float* __restrict__ nxyz_ws,
                                  int* __restrict__ idx_ws,
                                  float* __restrict__ out_idxf) {
  __shared__ float sx[BQ_CHUNK * 3];
  __shared__ int s_done;

  const int lane = threadIdx.x & 31;
  const int wid  = threadIdx.x >> 5;        // 0..7
  const int q    = blockIdx.x * 8 + wid;    // queries per block share batch b
  const int b    = (blockIdx.x * 8) / SS;
  const long qo  = (long)q * NS;

  if (threadIdx.x == 0) s_done = 0;

  const float qx = nxyz_ws[q * 3 + 0];
  const float qy = nxyz_ws[q * 3 + 1];
  const float qz = nxyz_ws[q * 3 + 2];

  int found = 0;
  int firstIdx = NN - 1;
  bool flagged = false;

  for (int cbase = 0; cbase < NN; cbase += BQ_CHUNK) {
    __syncthreads();
    if (s_done == 8) break;   // uniform after barrier
    // cooperative chunk load: global -> LDS
    const float* src = xyz + ((size_t)b * NN + cbase) * 3;
    for (int t = threadIdx.x; t < BQ_CHUNK * 3; t += blockDim.x) sx[t] = src[t];
    __syncthreads();

    if (found < NS) {
      for (int j0 = 0; j0 < BQ_CHUNK; j0 += 32) {
        const int jl = j0 + lane;
        const float dx = sx[jl * 3 + 0] - qx;
        const float dy = sx[jl * 3 + 1] - qy;
        const float dz = sx[jl * 3 + 2] - qz;
        const float d2 = dx * dx + dy * dy + dz * dz;
        const bool within = d2 < R2;
        const unsigned m = (unsigned)__ballot(within);
        if (m) {
          if (found == 0) firstIdx = cbase + j0 + (__ffs(m) - 1);
          if (within) {
            int pos = found + __popc(m & ((1u << lane) - 1u));
            if (pos < NS) {
              int jj = cbase + jl;
              idx_ws[qo + pos]  = jj;
              out_idxf[qo + pos] = (float)jj;
            }
          }
          found += __popc(m);
          if (found >= NS) break;
        }
      }
      if (found >= NS && !flagged) {
        flagged = true;
        if (lane == 0) atomicAdd(&s_done, 1);
      }
    }
  }

  // pad: repeat first hit; no hits at all -> N-1 (matches reference clamp)
  const int pad = (found == 0) ? (NN - 1) : firstIdx;
  const int start = found < NS ? found : NS;
  for (int p = start + lane; p < NS; p += 32) {
    idx_ws[qo + p]  = pad;
    out_idxf[qo + p] = (float)pad;
  }
}

// ---------------------------------------------------------------------------
// Kernel 3: grouping + 3-layer MLP (WMMA f16->f32) + max-pool over samples.
// blockDim = 128 (4 waves), one query per wave.
// Dynamic LDS layout (halves):
//   [0      .. 6143 ]  W0^T  (64 x 96)
//   [6144   .. 10239]  W1^T  (64 x 64)
//   [10240  .. 18431]  W2^T  (128 x 64)
//   [18432 + wid*6144] per-wave Xbuf (64 x 96)  — reused as Y1 (64 x 64)
//   [43008 + wid*4096] per-wave Ybuf (64 x 64)  — Y0
// ---------------------------------------------------------------------------
#define MLP_WAVES 4
#define MLP_LDS_HALVES (18432 + MLP_WAVES * 6144 + MLP_WAVES * 4096)

__global__ __launch_bounds__(MLP_WAVES * 32)
void mlp_maxpool_kernel(const float* __restrict__ xyz,
                        const float* __restrict__ feat,
                        const float* __restrict__ nxyz_ws,
                        const int* __restrict__ idx_ws,
                        const _Float16* __restrict__ gw0t,
                        const _Float16* __restrict__ gw1t,
                        const _Float16* __restrict__ gw2t,
                        const float* __restrict__ s0, const float* __restrict__ t0,
                        const float* __restrict__ s1, const float* __restrict__ t1,
                        const float* __restrict__ s2, const float* __restrict__ t2,
                        float* __restrict__ out_feat) {
  extern __shared__ __align__(32) _Float16 smem[];
  _Float16* w0t = smem;
  _Float16* w1t = smem + 6144;
  _Float16* w2t = smem + 10240;

  const int lane = threadIdx.x & 31;
  const int wid  = threadIdx.x >> 5;
  _Float16* Xbuf = smem + 18432 + wid * 6144;                    // 64 x 96
  _Float16* Ybuf = smem + 18432 + MLP_WAVES * 6144 + wid * 4096; // 64 x 64

  // stage weights once per workgroup
  for (int t = threadIdx.x; t < 6144; t += blockDim.x) w0t[t] = gw0t[t];
  for (int t = threadIdx.x; t < 4096; t += blockDim.x) w1t[t] = gw1t[t];
  for (int t = threadIdx.x; t < 8192; t += blockDim.x) w2t[t] = gw2t[t];
  __syncthreads();

  const int q = blockIdx.x * MLP_WAVES + wid;
  const int b = q / SS;
  const float qx = nxyz_ws[q * 3 + 0];
  const float qy = nxyz_ws[q * 3 + 1];
  const float qz = nxyz_ws[q * 3 + 2];
  const int* idxq = idx_ws + (size_t)q * NS;
  const float invR = 1.0f / RADIUS;

  // ---- group: build X (64 samples x [3 rel-xyz | 64 feat | 29 zero-pad]) ----
  for (int i = lane; i < NS; i += 32) {
    const int j = idxq[i];
    const float* px = xyz + ((size_t)b * NN + j) * 3;
    _Float16* row = Xbuf + i * 96;
    row[0] = (_Float16)((px[0] - qx) * invR);
    row[1] = (_Float16)((px[1] - qy) * invR);
    row[2] = (_Float16)((px[2] - qz) * invR);
    const float4* pf = (const float4*)(feat + ((size_t)b * NN + j) * CIN);
#pragma unroll
    for (int c = 0; c < CIN / 4; ++c) {
      float4 f = pf[c];
      row[3 + c * 4 + 0] = (_Float16)f.x;
      row[3 + c * 4 + 1] = (_Float16)f.y;
      row[3 + c * 4 + 2] = (_Float16)f.z;
      row[3 + c * 4 + 3] = (_Float16)f.w;
    }
#pragma unroll
    for (int k = 67; k < 96; ++k) row[k] = (_Float16)0.0f;
  }
  // per-wave private LDS tiles: same-wave DS ordering, no workgroup barrier needed

  const int chL = lane & 15;

  // ---- layer 0: X(64x96) @ W0 -> Y0(64x64), BN+ReLU ----
  for (int mt = 0; mt < 4; ++mt) {
    hvec16 a0 = load_a_frag(Xbuf, 96, mt, 0, lane);
    hvec16 a1 = load_a_frag(Xbuf, 96, mt, 1, lane);
    hvec16 a2 = load_a_frag(Xbuf, 96, mt, 2, lane);
#pragma unroll
    for (int nt = 0; nt < 4; ++nt) {
      fvec8 acc = {};
      acc = WMMA16(a0, load_b_frag(w0t, 96, nt, 0, lane), acc);
      acc = WMMA16(a1, load_b_frag(w0t, 96, nt, 1, lane), acc);
      acc = WMMA16(a2, load_b_frag(w0t, 96, nt, 2, lane), acc);
      const int ch = nt * 16 + chL;
      store_tile_relu(acc, s0[ch], t0[ch], Ybuf, 64, mt, nt, lane);
    }
  }

  // ---- layer 1: Y0(64x64) @ W1 -> Y1(64x64) into Xbuf, BN+ReLU ----
  for (int mt = 0; mt < 4; ++mt) {
    hvec16 a0 = load_a_frag(Ybuf, 64, mt, 0, lane);
    hvec16 a1 = load_a_frag(Ybuf, 64, mt, 1, lane);
#pragma unroll
    for (int nt = 0; nt < 4; ++nt) {
      fvec8 acc = {};
      acc = WMMA16(a0, load_b_frag(w1t, 64, nt, 0, lane), acc);
      acc = WMMA16(a1, load_b_frag(w1t, 64, nt, 1, lane), acc);
      const int ch = nt * 16 + chL;
      store_tile_relu(acc, s1[ch], t1[ch], Xbuf, 64, mt, nt, lane);
    }
  }

  // ---- layer 2: Y1(64x64) @ W2 -> (64x128), BN+ReLU, max over 64 rows ----
  float cmax[8], sc2[8], sh2[8];
#pragma unroll
  for (int nt = 0; nt < 8; ++nt) {
    const int ch = nt * 16 + chL;
    sc2[nt] = s2[ch];
    sh2[nt] = t2[ch];
    cmax[nt] = 0.0f;               // ReLU guarantees max >= 0
  }
  for (int mt = 0; mt < 4; ++mt) {
    hvec16 a0 = load_a_frag(Xbuf, 64, mt, 0, lane);
    hvec16 a1 = load_a_frag(Xbuf, 64, mt, 1, lane);
#pragma unroll
    for (int nt = 0; nt < 8; ++nt) {
      fvec8 acc = {};
      acc = WMMA16(a0, load_b_frag(w2t, 64, nt, 0, lane), acc);
      acc = WMMA16(a1, load_b_frag(w2t, 64, nt, 1, lane), acc);
#pragma unroll
      for (int r = 0; r < 8; ++r)
        cmax[nt] = fmaxf(cmax[nt], fmaf(acc[r], sc2[nt], sh2[nt]));
    }
  }
  // lane l holds rows {mt*16 + r + 8*(l>>4)}; partner rows live in lane l^16
#pragma unroll
  for (int nt = 0; nt < 8; ++nt) {
    float m = fmaxf(cmax[nt], __shfl_xor(cmax[nt], 16, 32));
    if (lane < 16) out_feat[(size_t)q * 128 + nt * 16 + lane] = m;
  }
}

// ---------------------------------------------------------------------------
// Host launcher
// ---------------------------------------------------------------------------
extern "C" void kernel_launch(void* const* d_in, const int* in_sizes, int n_in,
                              void* d_out, int out_size, void* d_ws, size_t ws_size,
                              hipStream_t stream) {
  (void)in_sizes; (void)n_in; (void)out_size; (void)ws_size;

  const float* xyz  = (const float*)d_in[0];   // (B,N,3)
  const float* feat = (const float*)d_in[1];   // (B,N,64)
  const int*   inds = (const int*)d_in[2];     // (B,S)
  const float* w0   = (const float*)d_in[3];
  const float* s0   = (const float*)d_in[4];
  const float* t0   = (const float*)d_in[5];
  const float* w1   = (const float*)d_in[6];
  const float* s1   = (const float*)d_in[7];
  const float* t1   = (const float*)d_in[8];
  const float* w2   = (const float*)d_in[9];
  const float* s2   = (const float*)d_in[10];
  const float* t2   = (const float*)d_in[11];

  // workspace layout (bytes)
  char* ws = (char*)d_ws;
  float*    nxyz_ws = (float*)ws;                         // B*S*3 f32   (196608 B)
  int*      idx_ws  = (int*)(ws + 196608);                // B*S*64 i32  (4 MiB)
  _Float16* w0t     = (_Float16*)(ws + 4390912);          // 64*96  f16
  _Float16* w1t     = (_Float16*)(ws + 4403200);          // 64*64  f16
  _Float16* w2t     = (_Float16*)(ws + 4411392);          // 128*64 f16

  // d_out layout (f32 elements), reference tuple order
  float* out_nxyz = (float*)d_out;                        // B*S*3
  float* out_feat = out_nxyz + (size_t)BB * SS * 3;       // B*S*128
  float* out_inds = out_feat + (size_t)BB * SS * 128;     // B*S
  float* out_idxf = out_inds + (size_t)BB * SS;           // B*S*64

  // 0) weights -> f16 transposed/padded
  prep_weights_kernel<<<72, 256, 0, stream>>>(w0, w1, w2, w0t, w1t, w2t);

  // 1) gather query points + inds output
  gather_setup_kernel<<<(BB * SS + 255) / 256, 256, 0, stream>>>(
      xyz, inds, nxyz_ws, out_nxyz, out_inds);

  // 2) ball query (8 queries per 256-thread block)
  ball_query_kernel<<<(BB * SS) / 8, 256, 0, stream>>>(
      xyz, nxyz_ws, idx_ws, out_idxf);

  // 3) group + WMMA MLP + max-pool (4 queries per 128-thread block)
  const size_t mlp_lds_bytes = (size_t)MLP_LDS_HALVES * sizeof(_Float16);
  mlp_maxpool_kernel<<<(BB * SS) / MLP_WAVES, MLP_WAVES * 32, mlp_lds_bytes, stream>>>(
      xyz, feat, nxyz_ws, idx_ws, w0t, w1t, w2t,
      s0, t0, s1, t1, s2, t2, out_feat);
}
